// ACDC_13134009991609
// MI455X (gfx1250) — matrix operand order repaired and checked by reading, not compiled, not run on hardware
//
#include <hip/hip_runtime.h>

// ============================================================================
// ACDC layer for MI455X (gfx1250):  out = riffle-IDCT( D ∘ DCT(A ∘ x) ) + bias
// Collapsed per 1024-wide group into out_seg = (x_seg ∘ A_seg) @ M_g + bias.
// M_g built on device via a small WMMA GEMM; GEMMs use 2-term bf16 split
// (hi*hi + lo*hi + hi*lo, f32 accumulate) on v_wmma_f32_16x16x32_bf16.
// Double-buffered K-pipeline, one barrier per K-step, LDS staging via
// GLOBAL_LOAD_ASYNC_TO_LDS_B128 (ASYNCcnt) when the builtin is available.
// Workspace layout (requires ws_size >= 36 MB):
//   [0,8M)   A1 hi   (4 groups x 1024x1024 bf16)   A1[g][j][k]=2cos(..)D[gk]
//   [8,16M)  A1 lo
//   [16,18M) A2t hi  (1024x1024 bf16)              A2t[j'][k]=Ci[perm(k)][j']
//   [18,20M) A2t lo
//   [20,28M) Mt hi   (4 x 1024x1024 bf16)          Mt[g][j'][j] = M_g^T
//   [28,36M) Mt lo
// ============================================================================

typedef __attribute__((ext_vector_type(16))) __bf16 v16bf;
typedef __attribute__((ext_vector_type(4)))  __bf16 v4bf;
typedef __attribute__((ext_vector_type(8)))  float  v8f;
typedef __attribute__((__vector_size__(16))) int    v4i_;

#if defined(__has_builtin)
#if __has_builtin(__builtin_amdgcn_global_load_async_to_lds_b128) && \
    __has_builtin(__builtin_amdgcn_s_wait_asynccnt)
#define USE_ASYNC 1
#endif
#endif
#ifndef USE_ASYNC
#define USE_ASYNC 0
#endif

#define AS_GLOBAL __attribute__((address_space(1)))
#define AS_SHARED __attribute__((address_space(3)))

union FragBF {
    uint4 q[2];
    v16bf v;
};

__device__ __forceinline__ void split2(float x, __bf16& h, __bf16& l) {
    h = (__bf16)x;
    l = (__bf16)(x - (float)h);
}

__device__ __forceinline__ void async_wait_all() {
#if USE_ASYNC
    __builtin_amdgcn_s_wait_asynccnt(0);
#endif
}

// ---------------------------------------------------------------------------
// Stage a 128x32 bf16 (hi,lo) tile pair, global (ld=1024) -> LDS (ld=32).
// Async path: issue GLOBAL_LOAD_ASYNC_TO_LDS_B128 (writes LDS directly, no
// VGPR data, tracked by ASYNCcnt). Sync fallback: prefetch to regs at issue,
// ds_store at commit (keeps global latency off the critical path).
// ---------------------------------------------------------------------------
__device__ __forceinline__ void stage_issue(uint4 regs[4],
                                            __bf16* __restrict__ dsthi,
                                            __bf16* __restrict__ dstlo,
                                            const __bf16* __restrict__ srchi,
                                            const __bf16* __restrict__ srclo,
                                            int row0, int k0, int tid) {
#pragma unroll
    for (int i = 0; i < 2; ++i) {
        int f   = tid * 2 + i;        // 0..511
        int row = f >> 2;             // 0..127
        int ch  = (f & 3) * 8;        // 0,8,16,24 (bf16 units)
        size_t go = (size_t)(row0 + row) * 1024 + (size_t)(k0 + ch);
        int    lo = row * 32 + ch;
#if USE_ASYNC
        (void)regs;
        __builtin_amdgcn_global_load_async_to_lds_b128(
            (AS_GLOBAL v4i_*)(srchi + go), (AS_SHARED v4i_*)(dsthi + lo), 0, 0);
        __builtin_amdgcn_global_load_async_to_lds_b128(
            (AS_GLOBAL v4i_*)(srclo + go), (AS_SHARED v4i_*)(dstlo + lo), 0, 0);
#else
        (void)dsthi; (void)dstlo;
        regs[i * 2 + 0] = *(const uint4*)(srchi + go);
        regs[i * 2 + 1] = *(const uint4*)(srclo + go);
#endif
    }
}

__device__ __forceinline__ void stage_commit(const uint4 regs[4],
                                             __bf16* __restrict__ dsthi,
                                             __bf16* __restrict__ dstlo, int tid) {
#if !USE_ASYNC
#pragma unroll
    for (int i = 0; i < 2; ++i) {
        int f   = tid * 2 + i;
        int row = f >> 2;
        int ch  = (f & 3) * 8;
        int lo  = row * 32 + ch;
        *(uint4*)(dsthi + lo) = regs[i * 2 + 0];
        *(uint4*)(dstlo + lo) = regs[i * 2 + 1];
    }
#else
    (void)regs; (void)dsthi; (void)dstlo; (void)tid;
#endif
}

// ---------------------------------------------------------------------------
// One K=32 step of the 128x128 block tile. Each wave owns a 32(M) x 64(N)
// region = 2x4 16x16 f32 accumulators. 3-term bf16-split accumulation.
// A-frag (16-bit A 16x32, ISA 7.12.2): lane<16 rows, elems0-7:K=0..7,
//   elems8-15:K=16..23; lanes>=16 shifted +8 in K.
// B-frag (16-bit B 32x16): col = lane&15, same K order as A-frag;
//   sB stored N-major (128 N rows, 32 K contiguous) -> 2x16B ds loads.
// ---------------------------------------------------------------------------
__device__ __forceinline__ void wmma_tiles(const __bf16* __restrict__ sAhi,
                                           const __bf16* __restrict__ sAlo,
                                           const __bf16* __restrict__ sBhi,
                                           const __bf16* __restrict__ sBlo,
                                           v8f acc[2][4], int wm, int wn, int lane) {
    const int l16  = lane & 15;
    const int half = lane >> 4;

    FragBF ah[2], al[2];
#pragma unroll
    for (int mi = 0; mi < 2; ++mi) {
        int row = wm * 32 + mi * 16 + l16;
        int o   = row * 32 + half * 8;
        ah[mi].q[0] = *(const uint4*)(sAhi + o);
        ah[mi].q[1] = *(const uint4*)(sAhi + o + 16);
        al[mi].q[0] = *(const uint4*)(sAlo + o);
        al[mi].q[1] = *(const uint4*)(sAlo + o + 16);
    }
#pragma unroll
    for (int ni = 0; ni < 4; ++ni) {
        int rn = wn * 64 + ni * 16 + l16;
        int o  = rn * 32 + half * 8;
        FragBF bh, bl;
        bh.q[0] = *(const uint4*)(sBhi + o);
        bh.q[1] = *(const uint4*)(sBhi + o + 16);
        bl.q[0] = *(const uint4*)(sBlo + o);
        bl.q[1] = *(const uint4*)(sBlo + o + 16);
#pragma unroll
        for (int mi = 0; mi < 2; ++mi) {
            acc[mi][ni] = __builtin_amdgcn_wmma_f32_16x16x32_bf16(
                false, ah[mi].v, false, bh.v, (short)0, acc[mi][ni], false, false);
            acc[mi][ni] = __builtin_amdgcn_wmma_f32_16x16x32_bf16(
                false, al[mi].v, false, bh.v, (short)0, acc[mi][ni], false, false);
            acc[mi][ni] = __builtin_amdgcn_wmma_f32_16x16x32_bf16(
                false, ah[mi].v, false, bl.v, (short)0, acc[mi][ni], false, false);
        }
    }
}

// ---------------------------------------------------------------------------
// Kernel 1: generate A1 (DCT * D, per group) and A2t (riffled inverse-DCT,
// transposed) analytically, split into bf16 hi/lo. cos args reduced mod the
// period (4096) so cosf stays accurate.
// ---------------------------------------------------------------------------
__global__ void __launch_bounds__(256)
acdc_genab(const float* __restrict__ Dv,
           __bf16* __restrict__ a1h, __bf16* __restrict__ a1l,
           __bf16* __restrict__ a2h, __bf16* __restrict__ a2l) {
    const float W = 3.14159265358979323846f / 2048.0f;
    size_t idx = (size_t)blockIdx.x * 256 + threadIdx.x;   // 5M total
    if (idx < (size_t)4 * 1024 * 1024) {
        int k = (int)(idx & 1023);
        int j = (int)((idx >> 10) & 1023);
        int g = (int)(idx >> 20);
        int t = (k * (2 * j + 1)) & 4095;
        float v = 2.0f * cosf(W * (float)t) * Dv[(g << 10) + k];
        __bf16 h, l; split2(v, h, l);
        a1h[idx] = h; a1l[idx] = l;
    } else {
        size_t i2 = idx - (size_t)4 * 1024 * 1024;
        int k  = (int)(i2 & 1023);
        int jp = (int)(i2 >> 10);
        int i  = ((k & 1) << 9) + (k >> 1);   // riffle^-1 of k
        float v;
        if (i == 0) v = 1.0f / 2048.0f;
        else {
            int t = (i * (2 * jp + 1)) & 4095;
            v = cosf(W * (float)t) * (1.0f / 1024.0f);
        }
        __bf16 h, l; split2(v, h, l);
        a2h[i2] = h; a2l[i2] = l;
    }
}

// ---------------------------------------------------------------------------
// Kernel 2: Mt_g = A2t @ A1_g^T (1024^3 per group, 3-term split), epilogue
// re-splits f32 into bf16 hi/lo. Double-buffered, async staging both sides.
// grid = (8, 8, 4), block = 256.
// ---------------------------------------------------------------------------
__global__ void __launch_bounds__(256)
acdc_build_mt(const __bf16* __restrict__ a2h, const __bf16* __restrict__ a2l,
              const __bf16* __restrict__ a1h, const __bf16* __restrict__ a1l,
              __bf16* __restrict__ mth, __bf16* __restrict__ mtl) {
    __shared__ __bf16 sAh[2][4096], sAl[2][4096], sBh[2][4096], sBl[2][4096];
    const int tid  = threadIdx.x;
    const int lane = tid & 31, wave = tid >> 5;
    const int wm = wave & 3, wn = wave >> 2;
    const int g  = blockIdx.z;
    const int m0 = blockIdx.y * 128;   // j' rows
    const int n0 = blockIdx.x * 128;   // j cols
    const size_t goff = (size_t)g << 20;
    const __bf16* gBh = a1h + goff;
    const __bf16* gBl = a1l + goff;

    uint4 ra[4], rb[4];
    stage_issue(ra, sAh[0], sAl[0], a2h, a2l, m0, 0, tid);
    stage_issue(rb, sBh[0], sBl[0], gBh, gBl, n0, 0, tid);
    stage_commit(ra, sAh[0], sAl[0], tid);
    stage_commit(rb, sBh[0], sBl[0], tid);
    async_wait_all();
    __syncthreads();

    v8f acc[2][4] = {};
    for (int kt = 0; kt < 32; ++kt) {
        const int  cur = kt & 1, nxt = cur ^ 1;
        const bool hn  = (kt + 1) < 32;
        if (hn) {   // issue into nxt: its last readers were before the barrier
            int k1 = (kt + 1) * 32;
            stage_issue(ra, sAh[nxt], sAl[nxt], a2h, a2l, m0, k1, tid);
            stage_issue(rb, sBh[nxt], sBl[nxt], gBh, gBl, n0, k1, tid);
        }
        wmma_tiles(sAh[cur], sAl[cur], sBh[cur], sBl[cur], acc, wm, wn, lane);
        if (hn) {
            stage_commit(ra, sAh[nxt], sAl[nxt], tid);
            stage_commit(rb, sBh[nxt], sBl[nxt], tid);
        }
        async_wait_all();
        __syncthreads();
    }

    const int l16 = lane & 15, half = lane >> 4;
#pragma unroll
    for (int mi = 0; mi < 2; ++mi) {
        int mB = m0 + wm * 32 + mi * 16 + half * 8;
#pragma unroll
        for (int ni = 0; ni < 4; ++ni) {
            int n = n0 + wn * 64 + ni * 16 + l16;
#pragma unroll
            for (int r = 0; r < 8; ++r) {
                __bf16 h, l; split2(acc[mi][ni][r], h, l);
                size_t o = goff + (size_t)(mB + r) * 1024 + n;
                mth[o] = h; mtl[o] = l;
            }
        }
    }
}

// ---------------------------------------------------------------------------
// Kernel 3 helpers: x-side register prefetch + f32 -> (bf16 hi, bf16 lo)
// split into LDS (A-operand, M-major, 128x32).
// ---------------------------------------------------------------------------
__device__ __forceinline__ void load_x(const float* __restrict__ x,
                                       const float* __restrict__ Asc,
                                       int cbase, int m0, int k0,
                                       int xrow, int xcol,
                                       float4& s4, float4 xr[4]) {
    s4 = *(const float4*)(Asc + cbase + k0 + xcol);
#pragma unroll
    for (int i = 0; i < 4; ++i) {
        int row = xrow + 32 * i;
        xr[i] = *(const float4*)(x + (size_t)(m0 + row) * 4096 + cbase + k0 + xcol);
    }
}

__device__ __forceinline__ void commit_x(__bf16* __restrict__ dsthi,
                                         __bf16* __restrict__ dstlo,
                                         int xrow, int xcol,
                                         float4 s4, const float4 xr[4]) {
#pragma unroll
    for (int i = 0; i < 4; ++i) {
        int row = xrow + 32 * i;
        __bf16 h0, l0, h1, l1, h2, l2, h3, l3;
        split2(xr[i].x * s4.x, h0, l0);
        split2(xr[i].y * s4.y, h1, l1);
        split2(xr[i].z * s4.z, h2, l2);
        split2(xr[i].w * s4.w, h3, l3);
        v4bf hv, lv;
        hv[0] = h0; hv[1] = h1; hv[2] = h2; hv[3] = h3;
        lv[0] = l0; lv[1] = l1; lv[2] = l2; lv[3] = l3;
        *(v4bf*)(dsthi + row * 32 + xcol) = hv;
        *(v4bf*)(dstlo + row * 32 + xcol) = lv;
    }
}

// ---------------------------------------------------------------------------
// Kernel 3: out[:, seg_g] = (x_seg ∘ A_seg) @ M_g + bias_g.
// Double-buffered; B-side async-to-LDS, A-side VALU split with reg prefetch.
// grid = (128, 8, 4), block = 256.
// ---------------------------------------------------------------------------
__global__ void __launch_bounds__(256)
acdc_main(const float* __restrict__ x, const float* __restrict__ Asc,
          const float* __restrict__ bias,
          const __bf16* __restrict__ mth, const __bf16* __restrict__ mtl,
          float* __restrict__ out) {
    __shared__ __bf16 sAh[2][4096], sAl[2][4096], sBh[2][4096], sBl[2][4096];
    const int tid  = threadIdx.x;
    const int lane = tid & 31, wave = tid >> 5;
    const int wm = wave & 3, wn = wave >> 2;
    const int m0 = blockIdx.x * 128;        // rows of x
    const int n0 = blockIdx.y * 128;        // output cols within group
    const int g  = blockIdx.z;
    const size_t goff = (size_t)g << 20;
    const int cbase = g << 10;
    const __bf16* gBh = mth + goff;
    const __bf16* gBl = mtl + goff;

    const int xrow = tid >> 3;              // 0..31
    const int xcol = (tid & 7) * 4;         // 0..28

    float4 s4, xr[4];
    uint4  rb[4];
    load_x(x, Asc, cbase, m0, 0, xrow, xcol, s4, xr);
    stage_issue(rb, sBh[0], sBl[0], gBh, gBl, n0, 0, tid);
    commit_x(sAh[0], sAl[0], xrow, xcol, s4, xr);
    stage_commit(rb, sBh[0], sBl[0], tid);
    async_wait_all();
    __syncthreads();

    v8f acc[2][4] = {};
    for (int kt = 0; kt < 32; ++kt) {
        const int  cur = kt & 1, nxt = cur ^ 1;
        const bool hn  = (kt + 1) < 32;
        if (hn) {
            int k1 = (kt + 1) * 32;
            load_x(x, Asc, cbase, m0, k1, xrow, xcol, s4, xr);
            stage_issue(rb, sBh[nxt], sBl[nxt], gBh, gBl, n0, k1, tid);
        }
        wmma_tiles(sAh[cur], sAl[cur], sBh[cur], sBl[cur], acc, wm, wn, lane);
        if (hn) {
            commit_x(sAh[nxt], sAl[nxt], xrow, xcol, s4, xr);
            stage_commit(rb, sBh[nxt], sBl[nxt], tid);
        }
        async_wait_all();
        __syncthreads();
    }

    const int l16 = lane & 15, half = lane >> 4;
#pragma unroll
    for (int ni = 0; ni < 4; ++ni) {
        int n = n0 + wn * 64 + ni * 16 + l16;
        float b = bias[cbase + n];
#pragma unroll
        for (int mi = 0; mi < 2; ++mi) {
            int mB = m0 + wm * 32 + mi * 16 + half * 8;
#pragma unroll
            for (int r = 0; r < 8; ++r)
                out[(size_t)(mB + r) * 4096 + cbase + n] = acc[mi][ni][r] + b;
        }
    }
}

// ---------------------------------------------------------------------------
extern "C" void kernel_launch(void* const* d_in, const int* in_sizes, int n_in,
                              void* d_out, int out_size, void* d_ws, size_t ws_size,
                              hipStream_t stream) {
    (void)in_sizes; (void)n_in; (void)out_size; (void)ws_size;
    const float* x    = (const float*)d_in[0];
    const float* A    = (const float*)d_in[1];
    const float* D    = (const float*)d_in[2];
    const float* bias = (const float*)d_in[3];
    float* out = (float*)d_out;

    char* ws = (char*)d_ws;                 // needs >= 36 MB
    const size_t MB = 1u << 20;
    __bf16* a1h = (__bf16*)(ws + 0 * MB);
    __bf16* a1l = (__bf16*)(ws + 8 * MB);
    __bf16* a2h = (__bf16*)(ws + 16 * MB);
    __bf16* a2l = (__bf16*)(ws + 18 * MB);
    __bf16* mth = (__bf16*)(ws + 20 * MB);
    __bf16* mtl = (__bf16*)(ws + 28 * MB);

    // 1) generate DCT/IDCT factor matrices (bf16 hi/lo): 5M elements
    acdc_genab<<<20480, 256, 0, stream>>>(D, a1h, a1l, a2h, a2l);
    // 2) fuse into per-group 1024x1024 Mt = (C·diag(D)·P·Cinv)^T
    acdc_build_mt<<<dim3(8, 8, 4), 256, 0, stream>>>(a2h, a2l, a1h, a1l, mth, mtl);
    // 3) main GEMM: 16384 x 1024 x 1024 per group
    acdc_main<<<dim3(128, 8, 4), 256, 0, stream>>>(x, A, bias, mth, mtl, out);
}
